// MultiHeadAttention_28295244546545
// MI455X (gfx1250) — matrix operand order repaired and checked by reading, not compiled, not run on hardware
//
#include <hip/hip_runtime.h>

// ---------------------------------------------------------------------------
// MI455X (gfx1250, wave32) fused multi-head attention
//   pass 1: weight convert f32->f16
//   pass 2: QKV projection via v_wmma_f32_16x16x32_f16
//   pass 3: flash attention, K/V tiles staged by the Tensor Data Mover
//           (tensor_load_to_lds + s_wait_tensorcnt, double buffered in LDS)
//   pass 4: output projection GEMM + bias via WMMA
// ---------------------------------------------------------------------------

#define B_ 2
#define S_ 2048
#define E_ 1024
#define H_ 16
#define D_ 64
#define TILE_ELEMS (32 * D_)   // one kv tile = 32 rows x 64 f16, contiguous 4KB

typedef __attribute__((ext_vector_type(16))) _Float16 v16h;
typedef __attribute__((ext_vector_type(8)))  _Float16 v8h;
typedef __attribute__((ext_vector_type(8)))  float    v8f;
typedef __attribute__((ext_vector_type(8)))  float    f32x8;
typedef unsigned int __attribute__((ext_vector_type(4))) u32x4;
typedef int  __attribute__((ext_vector_type(4))) i32x4;
typedef int  __attribute__((ext_vector_type(8))) i32x8;

#if defined(__has_builtin)
#if __has_builtin(__builtin_amdgcn_tensor_load_to_lds) && \
    __has_builtin(__builtin_amdgcn_s_wait_tensorcnt)
#define USE_TDM 1
#endif
#endif

#define CAT16(lo, hi) __builtin_shufflevector(lo, hi, 0,1,2,3,4,5,6,7,8,9,10,11,12,13,14,15)

static __device__ inline v8f wmma16(v16h a, v16h b, v8f c) {
    // D = A(16x32 f16) * B(32x16 f16) + C(16x16 f32)
    return __builtin_amdgcn_wmma_f32_16x16x32_f16(false, a, false, b, (short)0, c, false, false);
}

// Load A-fragment (16x32, 16-bit) from a row-major f16 row pointer.
// lane: M = lane%16; elems 0..15 -> K = kbase + (lane/16)*16 + e
static __device__ inline v16h load_a_frag(const _Float16* row, int kbase, int ph) {
    v8h lo = *(const v8h*)(row + kbase + ph * 16);
    v8h hi = *(const v8h*)(row + kbase + ph * 16 + 8);
    return CAT16(lo, hi);
}

#if defined(USE_TDM)
// TDM descriptor for a contiguous 4KB (2048 x f16) tile: 1-line tensor.
// D# group0: count=1 | lds_addr | global_addr[56:0] | type=2
// D# group1: data_size=2B, tensor_dim0 = tile_dim0 = 2048, tensor_dim1 = 1
static __device__ inline void tdm_load_tile_4kb(const _Float16* gsrc, unsigned lds_off) {
    const unsigned long long ga = (unsigned long long)(size_t)gsrc;
    u32x4 g0;
    g0.x = 1u;                                      // count=1 (valid user D#)
    g0.y = lds_off;                                 // LDS byte address
    g0.z = (unsigned)ga;                            // global_addr[31:0]
    g0.w = ((unsigned)(ga >> 32) & 0x01FFFFFFu)     // global_addr[56:32]
           | 0x80000000u;                           // type=2 ("image")
    i32x8 g1;
    g1[0] = 0x00010000;                 // workgroup_mask=0, data_size=1 (2 bytes)
    g1[1] = (int)(TILE_ELEMS << 16);    // tensor_dim0[15:0] in bits[31:16]
    g1[2] = (int)(1u << 16);            // tensor_dim0[31:16]=0 | tensor_dim1[15:0]=1
    g1[3] = (int)(TILE_ELEMS << 16);    // tensor_dim1[31:16]=0 | tile_dim0=2048
    g1[4] = 0;                          // tile_dim1=0, tile_dim2=0 (unused)
    g1[5] = TILE_ELEMS;                 // tensor_dim0_stride[31:0]
    g1[6] = 0;
    g1[7] = 0;
    const i32x4 z4 = {0, 0, 0, 0};
#if __clang_major__ >= 23
    const i32x8 z8 = {0, 0, 0, 0, 0, 0, 0, 0};
    __builtin_amdgcn_tensor_load_to_lds(g0, g1, z4, z4, z8, 0);
#else
    __builtin_amdgcn_tensor_load_to_lds(g0, g1, z4, z4, 0);
#endif
}
#endif

// ---------------------------------------------------------------------------
// Pass 1: convert weights to f16
// ---------------------------------------------------------------------------
__global__ void cvt_weights_kernel(const float* __restrict__ wq_f,
                                   const float* __restrict__ wo_f,
                                   _Float16* __restrict__ wq_h,
                                   _Float16* __restrict__ wo_h) {
    int i = blockIdx.x * blockDim.x + threadIdx.x;
    if (i < 3 * D_ * D_) wq_h[i] = (_Float16)wq_f[i];
    if (i < E_ * E_)     wo_h[i] = (_Float16)wo_f[i];
}

// ---------------------------------------------------------------------------
// Pass 2: QKV projection. One wave per (b,s): the 16 A-rows are the 16 heads.
//   qkv[b,s,h,e] = sum_d x[b,s,h*64+d] * W_qkv[e,d],  e in [0,192)
// Output layout (B,H,S,D) f16 so attention tiles are contiguous.
// ---------------------------------------------------------------------------
__global__ void __launch_bounds__(256)
qkv_proj_kernel(const float* __restrict__ x,
                const _Float16* __restrict__ wq,
                _Float16* __restrict__ Q,
                _Float16* __restrict__ K,
                _Float16* __restrict__ V) {
    const int lane = threadIdx.x & 31;
    const int wave = threadIdx.x >> 5;
    const int ph   = lane >> 4;      // lane-half
    const int ln   = lane & 15;

    const int tile = blockIdx.x * 8 + wave;   // 0 .. B*S-1
    const int b = tile / S_;
    const int s = tile % S_;

    // A-fragments: row m = head = ln, features d contiguous per head
    const float* xrow = x + (size_t)tile * E_ + (size_t)ln * D_;
    v16h a[2];
#pragma unroll
    for (int c = 0; c < 2; ++c) {
        f32x8 f0 = *(const f32x8*)(xrow + c * 32 + ph * 16);
        f32x8 f1 = *(const f32x8*)(xrow + c * 32 + ph * 16 + 8);
        v16h av;
#pragma unroll
        for (int e = 0; e < 8; ++e) {
            av[e]     = (_Float16)f0[e];
            av[8 + e] = (_Float16)f1[e];
        }
        a[c] = av;
    }

#pragma unroll
    for (int t = 0; t < 12; ++t) {
        // B-fragment: B[k=d][n] = W_qkv[t*16+n][d]; lane col n = ln, k contiguous
        const _Float16* wrow = wq + (size_t)(t * 16 + ln) * D_;
        v8f acc = {};
#pragma unroll
        for (int c = 0; c < 2; ++c) {
            v16h bf = *(const v16h*)(wrow + c * 32 + ph * 16);
            acc = wmma16(a[c], bf, acc);
        }
        _Float16* dst = (t < 4) ? Q : ((t < 8) ? K : V);
        const int d = (t & 3) * 16 + ln;
#pragma unroll
        for (int r = 0; r < 8; ++r) {
            const int hh = r + 8 * ph;   // C row = head
            dst[(((size_t)b * H_ + hh) * S_ + s) * D_ + d] = (_Float16)acc[r];
        }
    }
}

// ---------------------------------------------------------------------------
// Pass 3: flash attention. Block = 8 waves = 128 queries of one (b,h).
// K/V kv-tiles (4KB each) staged into double-buffered LDS by the TDM;
// wave 0 issues, waits TENSORcnt, barrier publishes to all waves.
// ---------------------------------------------------------------------------
__global__ void __launch_bounds__(256)
attn_kernel(const _Float16* __restrict__ Q,
            const _Float16* __restrict__ K,
            const _Float16* __restrict__ V,
            _Float16* __restrict__ ctx) {
    __shared__ _Float16 Kbuf[2][TILE_ELEMS];   // kv tile of K, row-major
    __shared__ _Float16 Vraw[2][TILE_ELEMS];   // kv tile of V, row-major
    __shared__ _Float16 Vt[64][32];            // V tile transposed: Vt[d][kv_local]
    __shared__ _Float16 Pbuf[8][16][32];       // per-wave P tile (C->A relayout)

    const int lane = threadIdx.x & 31;
    const int wave = threadIdx.x >> 5;
    const int ph   = lane >> 4;
    const int ln   = lane & 15;

    const int QB = S_ / 128;                       // 16 query blocks per (b,h)
    const int qb = blockIdx.x % QB;
    const int h  = (blockIdx.x / QB) % H_;
    const int b  = blockIdx.x / (QB * H_);

    const _Float16* Qb = Q + ((size_t)b * H_ + h) * S_ * D_;
    const _Float16* Kb = K + ((size_t)b * H_ + h) * S_ * D_;
    const _Float16* Vb = V + ((size_t)b * H_ + h) * S_ * D_;

    const int q0 = qb * 128 + wave * 16;

    // Q A-fragments, resident for the whole kv loop
    v16h aq[2];
    {
        const _Float16* qrow = Qb + (size_t)(q0 + ln) * D_;
#pragma unroll
        for (int c = 0; c < 2; ++c) aq[c] = load_a_frag(qrow, c * 32, ph);
    }

    v8f o[4] = {};
    float m_run[8], l_run[8];
#pragma unroll
    for (int r = 0; r < 8; ++r) { m_run[r] = -3.0e38f; l_run[r] = 0.0f; }

    // ---- stage tile 0 ----
#if defined(USE_TDM)
    const unsigned koff0 = (unsigned)(size_t)&Kbuf[0][0];
    const unsigned koff1 = (unsigned)(size_t)&Kbuf[1][0];
    const unsigned voff0 = (unsigned)(size_t)&Vraw[0][0];
    const unsigned voff1 = (unsigned)(size_t)&Vraw[1][0];
    if (wave == 0) {
        tdm_load_tile_4kb(Kb, koff0);
        tdm_load_tile_4kb(Vb, voff0);
    }
#else
    {
        const int t = threadIdx.x;
        *(v8h*)&Kbuf[0][t * 8] = *(const v8h*)(Kb + t * 8);
        *(v8h*)&Vraw[0][t * 8] = *(const v8h*)(Vb + t * 8);
    }
#endif

    for (int kv = 0; kv < S_; kv += 32) {
        const int cur = (kv >> 5) & 1;
#if defined(USE_TDM)
        if (wave == 0) __builtin_amdgcn_s_wait_tensorcnt(0);
#endif
        __syncthreads();   // K/V tile `cur` ready; prior readers of Vt retired

        // --- transpose V tile (LDS -> LDS): Vt[d][kv_local] ---
        {
            const int idx = threadIdx.x * 8;       // 256 threads x 8 = 2048 elems
            const int r32 = idx >> 6;              // kv row 0..31
            const int d0  = idx & 63;
            v8h vv = *(const v8h*)(&Vraw[cur][r32 * D_ + d0]);
#pragma unroll
            for (int e = 0; e < 8; ++e) Vt[d0 + e][r32] = vv[e];
        }

        // --- kick off DMA for the next kv tile into the other buffer ---
        if (kv + 32 < S_) {
#if defined(USE_TDM)
            if (wave == 0) {
                tdm_load_tile_4kb(Kb + (size_t)(kv + 32) * D_, cur ? koff0 : koff1);
                tdm_load_tile_4kb(Vb + (size_t)(kv + 32) * D_, cur ? voff0 : voff1);
            }
#else
            const int t = threadIdx.x;
            __builtin_prefetch((const void*)(Kb + (size_t)(kv + 64) * D_ + t * 8), 0, 0);
            *(v8h*)&Kbuf[cur ^ 1][t * 8] = *(const v8h*)(Kb + (size_t)(kv + 32) * D_ + t * 8);
            *(v8h*)&Vraw[cur ^ 1][t * 8] = *(const v8h*)(Vb + (size_t)(kv + 32) * D_ + t * 8);
#endif
        }
        __syncthreads();   // Vt published

        // --- scores: S = Q K^T for two 16-col sub-tiles, K frags from LDS ---
        v8f c0 = {}, c1 = {};
#pragma unroll
        for (int c = 0; c < 2; ++c) {
            v16h bk0 = *(const v16h*)(&Kbuf[cur][ln * D_ + c * 32 + ph * 16]);
            v16h bk1 = *(const v16h*)(&Kbuf[cur][(16 + ln) * D_ + c * 32 + ph * 16]);
            c0 = wmma16(aq[c], bk0, c0);
            c1 = wmma16(aq[c], bk1, c1);
        }

        // --- online softmax (rows r+8*ph, cross-lane over 16-lane half) ---
#pragma unroll
        for (int r = 0; r < 8; ++r) {
            float s0 = c0[r] * 0.125f;             // 1/sqrt(64)
            float s1 = c1[r] * 0.125f;
            float vmax = fmaxf(s0, s1);
#pragma unroll
            for (int mm = 8; mm >= 1; mm >>= 1)
                vmax = fmaxf(vmax, __shfl_xor(vmax, mm, 32));
            const float mnew  = fmaxf(m_run[r], vmax);
            const float p0    = __expf(s0 - mnew);
            const float p1    = __expf(s1 - mnew);
            const float alpha = __expf(m_run[r] - mnew);
            float psum = p0 + p1;
#pragma unroll
            for (int mm = 8; mm >= 1; mm >>= 1)
                psum += __shfl_xor(psum, mm, 32);
            l_run[r] = l_run[r] * alpha + psum;
            m_run[r] = mnew;
#pragma unroll
            for (int j = 0; j < 4; ++j) o[j][r] *= alpha;
            const int rr = r + 8 * ph;             // C row
            Pbuf[wave][rr][ln]      = (_Float16)p0;
            Pbuf[wave][rr][16 + ln] = (_Float16)p1;
        }

        // --- P (C-layout in LDS) reloaded as A-fragment, wave-local ---
        v16h pa = load_a_frag(&Pbuf[wave][ln][0], 0, ph);

        // --- O += P * V : B-frags from transposed V in LDS ---
#pragma unroll
        for (int j = 0; j < 4; ++j) {
            v16h bv = *(const v16h*)(&Vt[j * 16 + ln][ph * 16]);
            o[j] = wmma16(pa, bv, o[j]);
        }
        // no trailing barrier needed: next iteration's first barrier retires
        // all reads of Vt / Kbuf[cur] before they are overwritten
    }

    // --- epilogue: normalize and store ctx as (B,S,E) f16 ---
#pragma unroll
    for (int j = 0; j < 4; ++j) {
#pragma unroll
        for (int r = 0; r < 8; ++r) {
            const int sq = q0 + r + 8 * ph;
            const float val = o[j][r] / l_run[r];
            ctx[((size_t)(b * S_ + sq)) * E_ + h * D_ + j * 16 + ln] = (_Float16)val;
        }
    }
}

// ---------------------------------------------------------------------------
// Pass 4: out = ctx (4096 x 1024) * W_out^T + b_out. One wave: 16x64 strip.
// ---------------------------------------------------------------------------
__global__ void __launch_bounds__(256)
out_proj_kernel(const _Float16* __restrict__ ctx,
                const _Float16* __restrict__ wo,
                const float* __restrict__ b_out,
                float* __restrict__ out) {
    const int lane = threadIdx.x & 31;
    const int wave = threadIdx.x >> 5;
    const int ph   = lane >> 4;
    const int ln   = lane & 15;

    const int g  = blockIdx.x * 8 + wave;     // 4096 waves
    const int rt = g >> 4;                    // row tile 0..255 (16 rows each)
    const int cs = g & 15;                    // 64-col strip 0..15

    const _Float16* arow = ctx + (size_t)(rt * 16 + ln) * E_;
    v8f acc[4] = {};
    for (int kc = 0; kc < E_ / 32; ++kc) {
        v16h a = load_a_frag(arow, kc * 32, ph);
#pragma unroll
        for (int j = 0; j < 4; ++j) {
            const int oc = cs * 64 + j * 16 + ln;
            // B[k=i][n=o] = W_out[o][i], k contiguous in i
            v16h bf = *(const v16h*)(wo + (size_t)oc * E_ + kc * 32 + ph * 16);
            acc[j] = wmma16(a, bf, acc[j]);
        }
    }
#pragma unroll
    for (int j = 0; j < 4; ++j) {
        const int oc = cs * 64 + j * 16 + ln;
        const float bias = b_out[oc];
#pragma unroll
        for (int r = 0; r < 8; ++r) {
            const int row = rt * 16 + r + 8 * ph;
            out[(size_t)row * E_ + oc] = acc[j][r] + bias;
        }
    }
}

// ---------------------------------------------------------------------------
extern "C" void kernel_launch(void* const* d_in, const int* in_sizes, int n_in,
                              void* d_out, int out_size, void* d_ws, size_t ws_size,
                              hipStream_t stream) {
    const float* x     = (const float*)d_in[0];
    const float* W_qkv = (const float*)d_in[1];
    const float* W_out = (const float*)d_in[2];
    const float* b_out = (const float*)d_in[3];
    float* out = (float*)d_out;

    // Workspace layout (f16 elements)
    const size_t nQ = (size_t)B_ * H_ * S_ * D_;   // 4,194,304
    _Float16* ws  = (_Float16*)d_ws;
    _Float16* Qh  = ws;
    _Float16* Kh  = Qh + nQ;
    _Float16* Vh  = Kh + nQ;
    _Float16* ctx = Vh + nQ;                       // (B,S,E)
    _Float16* wqh = ctx + nQ;                      // 3*D*D = 12288
    _Float16* woh = wqh + 3 * D_ * D_;             // E*E

    // 1) weight conversion
    cvt_weights_kernel<<<(E_ * E_ + 255) / 256, 256, 0, stream>>>(W_qkv, W_out, wqh, woh);

    // 2) QKV projection: B*S = 4096 waves -> 512 blocks of 8 waves
    qkv_proj_kernel<<<(B_ * S_) / 8, 256, 0, stream>>>(x, wqh, Qh, Kh, Vh);

    // 3) flash attention: B*H*(S/128) = 512 blocks
    attn_kernel<<<B_ * H_ * (S_ / 128), 256, 0, stream>>>(Qh, Kh, Vh, ctx);

    // 4) output projection: (B*S/16)*(E/64) = 4096 waves -> 512 blocks
    out_proj_kernel<<<((B_ * S_ / 16) * (E_ / 64)) / 8, 256, 0, stream>>>(ctx, woh, b_out, out);
}
